// DTVNet_88630945120923
// MI455X (gfx1250) — matrix operand order repaired and checked by reading, not compiled, not run on hardware
//
#include <hip/hip_runtime.h>

#define D_ 64
#define H_ 128
#define W_ 128
#define V_ (D_*H_*W_)   // 1048576
#define HW_ (H_*W_)     // 16384

typedef __attribute__((ext_vector_type(16))) __bf16 v16bf;
typedef __attribute__((ext_vector_type(8)))  float  v8f;
typedef __attribute__((ext_vector_type(4)))  int    v4i;
typedef __attribute__((ext_vector_type(8)))  int    v8i;

union BF16x16 { v16bf v; unsigned short u[16]; uint4 q[2]; };
union F32x8   { v8f v; float f[8]; };
union BF16x8  { uint4 q; unsigned short u[8]; };

__device__ __forceinline__ unsigned short f2bf(float f) {
  unsigned int u = __builtin_bit_cast(unsigned int, f);
  unsigned int r = u + 0x7FFFu + ((u >> 16) & 1u);   // round-to-nearest-even
  return (unsigned short)(r >> 16);
}
__device__ __forceinline__ float bf2f(unsigned short s) {
  unsigned int u = ((unsigned int)s) << 16;
  return __builtin_bit_cast(float, u);
}

// ---------------- weight packing ----------------
// w1p[(i*16+oc)*32 + tap]  (tap 27..31 zero), bf16
__global__ void pack_w1_k(const float* __restrict__ w1, unsigned short* __restrict__ w1p) {
  int idx = blockIdx.x*256 + threadIdx.x;
  if (idx >= 4*16*32) return;
  int tap = idx & 31, ioc = idx >> 5;
  w1p[idx] = (tap < 27) ? f2bf(w1[ioc*27 + tap]) : (unsigned short)0;
}
// w2p[((i*28+tap)*16+oc)*16+ic]  (tap 27 zero), bf16
__global__ void pack_w2_k(const float* __restrict__ w2, unsigned short* __restrict__ w2p) {
  int idx = blockIdx.x*256 + threadIdx.x;
  if (idx >= 4*28*256) return;
  int ic = idx & 15, oc = (idx >> 4) & 15, tp = (idx >> 8) % 28, i = (idx >> 8) / 28;
  unsigned short v = 0;
  if (tp < 27) v = f2bf(w2[((i*16 + oc)*16 + ic)*27 + tp]);
  w2p[idx] = v;
}

// ---------------- elementwise / reduction helpers ----------------
__global__ void zero_k(float* __restrict__ p, int n) {
  int i = blockIdx.x*256 + threadIdx.x;
  if (i < n) p[i] = 0.f;
}
__global__ void copy_k(const float* __restrict__ src, float* __restrict__ dst) {
  int i = blockIdx.x*256 + threadIdx.x;
  dst[i] = src[i];
}
__global__ void reduce_r_k(const float* __restrict__ t, const float* __restrict__ sino,
                           float* __restrict__ r) {
  int hw = blockIdx.x*256 + threadIdx.x;
  float s = 0.f;
  #pragma unroll 4
  for (int d = 0; d < D_; ++d) s += t[d*HW_ + hw];
  r[hw] = (sino[hw] - s) * (1.0f/(float)D_);
}
__global__ void make_z_k(const float* __restrict__ t, const float* __restrict__ r,
                         float* __restrict__ z) {
  int i = blockIdx.x*256 + threadIdx.x;
  z[i] = t[i] + r[i & (HW_-1)];
}
__global__ void fdiff_k(const float* __restrict__ z, float* __restrict__ g, int shift, int dim) {
  int i = blockIdx.x*256 + threadIdx.x;
  int c = (i >> shift) & (dim - 1);
  g[i] = (c < dim-1) ? (z[i + (1 << shift)] - z[i]) : 0.f;
}

// ---------------- conv1: 1ch -> 16ch, async-LDS staged, 1 WMMA per tile ------------
// One block = one (d,h) row. 9 halo rows of g staged to LDS (f32), halo pre-zeroed.
#define C1ROWE 136     // f32 elems per LDS row: halo at elem 3 and 132, main at 4..131

__global__ void __launch_bounds__(256) conv1_wmma_k(const float* __restrict__ g,
    const unsigned short* __restrict__ w1p, const float* __restrict__ b1,
    int ae, unsigned short* __restrict__ h1) {
  __shared__ __align__(16) float lg[9*C1ROWE];
  int hh = blockIdx.x & (H_-1);
  int dd = blockIdx.x >> 7;

  unsigned lbase = (unsigned)(uintptr_t)(void*)lg;
  // stage 9 rows x 512B = 288 chunks of 16B
  for (int cidx = threadIdx.x; cidx < 288; cidx += 256) {
    int ri = cidx >> 5, ch = cidx & 31;
    int od = ri/3 - 1, oh = ri - (ri/3)*3 - 1;
    int d2 = dd + od, h2v = hh + oh;
    if ((unsigned)d2 < D_ && (unsigned)h2v < H_) {
      unsigned dst = lbase + (unsigned)((ri*C1ROWE + 4)*4 + ch*16);
      const void* src = (const void*)(g + (size_t)(d2*H_ + h2v)*W_ + ch*4);
      asm volatile("global_load_async_to_lds_b128 %0, %1, off"
                   :: "v"(dst), "v"(src) : "memory");
    } else {
      *(uint4*)((char*)lg + (ri*C1ROWE + 4)*4 + ch*16) = make_uint4(0,0,0,0);
    }
  }
  if (threadIdx.x < 18) {               // zero halo voxels
    int ri = threadIdx.x >> 1;
    int e = (threadIdx.x & 1) ? 132 : 3;
    lg[ri*C1ROWE + e] = 0.f;
  }
  asm volatile("s_wait_asynccnt 0x0" ::: "memory");
  __syncthreads();

  int lane = threadIdx.x & 31;
  int wave = threadIdx.x >> 5;
  int n = lane & 15, half = lane >> 4;
  int w0 = wave*16 + n;

  BF16x16 a;   // row=oc, K=tap (27 real + 5 zero)
  const unsigned short* wb = w1p + (ae*16 + n)*32;
  a.q[0] = *(const uint4*)(wb + half*16);
  a.q[1] = *(const uint4*)(wb + half*16 + 8);

  BF16x16 b;   // elem e -> K = 16*half + e (tap index), voxel column n
  #pragma unroll
  for (int e = 0; e < 16; ++e) {
    int tap = 16*half + e;
    unsigned short val = 0;
    if (tap < 27) {
      int od = tap/9, oh = (tap/3)%3, ow = tap%3;     // 0..2
      int ri = od*3 + oh;
      val = f2bf(lg[ri*C1ROWE + 4 + (w0 - 1 + ow)]);  // elem in [3,132]
    }
    b.u[e] = val;
  }

  F32x8 c;
  #pragma unroll
  for (int r = 0; r < 8; ++r) c.f[r] = b1[ae*16 + 8*half + r];   // M = r + 8*half

  c.v = __builtin_amdgcn_wmma_f32_16x16x32_bf16(false, a.v, false, b.v, (short)0, c.v, false, false);

  BF16x8 o;
  #pragma unroll
  for (int r = 0; r < 8; ++r) { float f = c.f[r]; o.u[r] = f2bf(f > 0.f ? f : 0.f); }
  *(uint4*)(h1 + (size_t)((dd*H_ + hh)*W_ + w0)*16 + 8*half) = o.q;
}

// ---------------- conv2: 16ch -> 16ch, TDM weights + async rows, 14 WMMAs ----------
#define ROWV 130                 // 1 + 128 + 1 halo voxels
#define ROWE (ROWV*16)           // ushort elements per LDS row (2080)
#define ROWB (ROWE*2)            // bytes per LDS row (4160)

__global__ void __launch_bounds__(256) conv2_wmma_k(const unsigned short* __restrict__ h1,
    const unsigned short* __restrict__ w2p, const float* __restrict__ b2,
    int ae, unsigned short* __restrict__ h2) {
  __shared__ __align__(16) unsigned short lw[28*256];      // packed weights (14336B)
  __shared__ __align__(16) unsigned short lbuf[9*ROWE];    // 9 halo-padded input rows

  int hh = blockIdx.x & (H_-1);
  int dd = blockIdx.x >> 7;

  // ---- stage weights with the Tensor Data Mover (wave 0 issues one 1-D D# load) ----
  if ((threadIdx.x >> 5) == 0) {
    unsigned lwoff = (unsigned)(uintptr_t)(void*)lw;
    unsigned long long ga = (unsigned long long)(uintptr_t)(const void*)(w2p + (size_t)ae*28*256);
    v4i g0; v8i g1; v4i g2; v4i g3;
    g0[0] = 1;                                   // count=1 valid descriptor
    g0[1] = (int)lwoff;                          // lds_addr (bytes)
    g0[2] = (int)(unsigned)(ga & 0xFFFFFFFFu);   // global_addr[31:0]
    g0[3] = (int)((unsigned)(ga >> 32) | 0x80000000u); // global_addr[56:32] | type=2
    g1[0] = 3 << 16;                             // data_size=3 (8B units)
    g1[1] = (int)(1792u << 16);                  // tensor_dim0[15:0] = 1792 (14336B/8)
    g1[2] = (int)(1u << 16);                     // tensor_dim0 hi=0; tensor_dim1=1
    g1[3] = (int)(1792u << 16);                  // tensor_dim1 hi=0; tile_dim0=1792
    g1[4] = 1;                                   // tile_dim1=1, tile_dim2=0 (unused)
    g1[5] = 1792;                                // tensor_dim0_stride
    g1[6] = 0;
    g1[7] = 0;
    g2[0] = 1; g2[1] = 1; g2[2] = 0; g2[3] = 0;  // tensor_dim2/3 = 1, tile_dim3=0
    g3[0] = 0; g3[1] = 0; g3[2] = 0; g3[3] = 0;
    asm volatile("tensor_load_to_lds %0, %1, %2, %3"
                 :: "s"(g0), "s"(g1), "s"(g2), "s"(g3) : "memory");
  }

  // ---- stage 9 input rows (4096B each) via async LDS DMA; OOB rows zero-filled ----
  unsigned lbase = (unsigned)(uintptr_t)(void*)lbuf;
  #pragma unroll
  for (int ri = 0; ri < 9; ++ri) {
    int od = ri/3 - 1, oh = ri - (ri/3)*3 - 1;
    int d2 = dd + od, h2v = hh + oh;
    if ((unsigned)d2 < D_ && (unsigned)h2v < H_) {         // block-uniform predicate
      unsigned dst = lbase + ri*ROWB + 32 + threadIdx.x*16;
      const void* src = (const void*)((const char*)(h1 + (size_t)((d2*H_ + h2v)*W_)*16)
                                      + threadIdx.x*16);
      asm volatile("global_load_async_to_lds_b128 %0, %1, off"
                   :: "v"(dst), "v"(src) : "memory");
    } else {
      *(uint4*)((char*)lbuf + ri*ROWB + 32 + threadIdx.x*16) = make_uint4(0,0,0,0);
    }
  }
  if (threadIdx.x < 36) {   // zero the W halos (32B each side per row)
    int ri = threadIdx.x >> 2, wch = threadIdx.x & 3;
    int off = ri*ROWB + (wch < 2 ? wch*16 : 4128 + (wch - 2)*16);
    *(uint4*)((char*)lbuf + off) = make_uint4(0,0,0,0);
  }
  asm volatile("s_wait_asynccnt 0x0" ::: "memory");
  __builtin_amdgcn_s_wait_tensorcnt(0);
  __syncthreads();

  int lane = threadIdx.x & 31;
  int wave = threadIdx.x >> 5;                // tile index along W
  int n = lane & 15, half = lane >> 4;
  int w0 = wave*16 + n;

  F32x8 c;
  #pragma unroll
  for (int r = 0; r < 8; ++r) c.f[r] = b2[ae*16 + 8*half + r];

  #pragma unroll
  for (int s = 0; s < 14; ++s) {              // K step: taps (2s, 2s+1); tap 27 zero weights
    int j0 = 2*s, j1 = 2*s + 1;
    BF16x16 a;
    a.q[0] = *(const uint4*)(lw + ((2*s + half)*16 + n)*16);
    a.q[1] = *(const uint4*)(lw + ((2*s + half)*16 + n)*16 + 8);

    int jb = 2*s + half; if (jb > 26) jb = 26;   // finite values vs zero weights
    int od = jb/9, oh = (jb/3)%3, ow = jb%3;     // 0..2 each
    int ri = od*3 + oh;
    BF16x16 b;
    const uint4* src = (const uint4*)(lbuf + ri*ROWE + (w0 + ow)*16);
    b.q[0] = src[0]; b.q[1] = src[1];
    c.v = __builtin_amdgcn_wmma_f32_16x16x32_bf16(false, a.v, false, b.v, (short)0, c.v, false, false);
  }

  BF16x8 o;
  #pragma unroll
  for (int r = 0; r < 8; ++r) { float f = c.f[r]; o.u[r] = f2bf(f > 0.f ? f : 0.f); }
  *(uint4*)(h2 + (size_t)((dd*H_ + hh)*W_ + w0)*16 + 8*half) = o.q;
}

// ---------------- conv3: 16ch -> 1ch + residual ----------------
__global__ void conv3_res_k(const unsigned short* __restrict__ h2,
    const float* __restrict__ w3, const float* __restrict__ b3,
    const float* __restrict__ resid, int ae, float* __restrict__ out) {
  int i = blockIdx.x*256 + threadIdx.x;
  int w = i & (W_-1), hh = (i >> 7) & (H_-1), dd = i >> 14;
  float acc = b3[ae] + resid[i];
  const float* wv = w3 + ae*16*27;
  for (int kd = -1; kd <= 1; ++kd)
    for (int kh = -1; kh <= 1; ++kh)
      for (int kw = -1; kw <= 1; ++kw) {
        int d2 = dd+kd, h2v = hh+kh, w2v = w+kw;
        if ((unsigned)d2 < D_ && (unsigned)h2v < H_ && (unsigned)w2v < W_) {
          int tap = (kd+1)*9 + (kh+1)*3 + (kw+1);
          const unsigned short* hv = h2 + (size_t)((d2*H_ + h2v)*W_ + w2v)*16;
          #pragma unroll
          for (int ic = 0; ic < 16; ++ic)
            acc += bf2f(hv[ic]) * wv[ic*27 + tap];
        }
      }
  out[i] = acc;
}

// ---------------- cascade state updates ----------------
__global__ void update_pqs_k(float* __restrict__ p, float* __restrict__ q, float* __restrict__ s,
    const float* __restrict__ pn, const float* __restrict__ qn, const float* __restrict__ sn,
    const float* __restrict__ ntx, const float* __restrict__ nty, const float* __restrict__ ntz,
    int c) {
  int i = blockIdx.x*256 + threadIdx.x;
  float a = ntx[c], b = nty[c], g = ntz[c];
  p[i] = (1.f+a)*p[i] - a*pn[i];
  q[i] = (1.f+b)*q[i] - b*qn[i];
  s[i] = (1.f+g)*s[i] - g*sn[i];
}
__global__ void combine_k(const float* __restrict__ p, const float* __restrict__ q,
    const float* __restrict__ s, const float* __restrict__ t_old,
    const float* __restrict__ zn, const float* __restrict__ nt, int c,
    float* __restrict__ t_out) {
  int i = blockIdx.x*256 + threadIdx.x;
  int w = i & (W_-1), hh = (i >> 7) & (H_-1), dd = i >> 14;
  float a = nt[c];
  float acc = t_old[i] + a*(t_old[i] - zn[i]);
  acc += (w  > 0 ? p[i-1]   : 0.f) - (w  < W_-1 ? p[i] : 0.f);
  acc += (hh > 0 ? q[i-W_]  : 0.f) - (hh < H_-1 ? q[i] : 0.f);
  acc += (dd > 0 ? s[i-HW_] : 0.f) - (dd < D_-1 ? s[i] : 0.f);
  t_out[i] = acc;
}

extern "C" void kernel_launch(void* const* d_in, const int* in_sizes, int n_in,
                              void* d_out, int out_size, void* d_ws, size_t ws_size,
                              hipStream_t stream) {
  const float* image = (const float*)d_in[0];
  const float* sino  = (const float*)d_in[1];
  const float* w1 = (const float*)d_in[2];
  const float* b1 = (const float*)d_in[3];
  const float* w2 = (const float*)d_in[4];
  const float* b2 = (const float*)d_in[5];
  const float* w3 = (const float*)d_in[6];
  const float* b3 = (const float*)d_in[7];
  const float* ntx = (const float*)d_in[8];
  const float* nty = (const float*)d_in[9];
  const float* ntz = (const float*)d_in[10];
  const float* ntt = (const float*)d_in[11];
  float* out = (float*)d_out;

  float* ws = (float*)d_ws;
  float* P  = ws;          float* Q  = ws + 1*V_;  float* S  = ws + 2*V_;
  float* Z  = ws + 3*V_;   float* G  = ws + 4*V_;
  float* PN = ws + 5*V_;   float* QN = ws + 6*V_;  float* SN = ws + 7*V_;  float* ZN = ws + 8*V_;
  float* R  = ws + 9*V_;                                   // HW_ floats
  unsigned short* H1  = (unsigned short*)(ws + 9*V_ + HW_);
  unsigned short* H2  = H1 + (size_t)16*V_;
  unsigned short* W1P = H2 + (size_t)16*V_;
  unsigned short* W2P = W1P + 4*16*32;

  dim3 blk(256);
  pack_w1_k<<<dim3((4*16*32 + 255)/256), blk, 0, stream>>>(w1, W1P);
  pack_w2_k<<<dim3((4*28*256 + 255)/256), blk, 0, stream>>>(w2, W2P);
  zero_k<<<dim3((3*V_)/256), blk, 0, stream>>>(P, 3*V_);
  copy_k<<<dim3(V_/256), blk, 0, stream>>>(image, out);     // outs[0] = image

  float* nouts[4] = {PN, QN, SN, ZN};
  for (int c = 0; c < 3; ++c) {
    const float* t_in = out + (size_t)c*V_;
    float* t_out = out + (size_t)(c+1)*V_;
    reduce_r_k<<<dim3(HW_/256), blk, 0, stream>>>(t_in, sino, R);
    make_z_k<<<dim3(V_/256), blk, 0, stream>>>(t_in, R, Z);
    for (int b = 0; b < 4; ++b) {
      const float* gin;
      if (b == 0)      { fdiff_k<<<dim3(V_/256), blk, 0, stream>>>(Z, G, 0,  W_); gin = G; }
      else if (b == 1) { fdiff_k<<<dim3(V_/256), blk, 0, stream>>>(Z, G, 7,  H_); gin = G; }
      else if (b == 2) { fdiff_k<<<dim3(V_/256), blk, 0, stream>>>(Z, G, 14, D_); gin = G; }
      else             { gin = Z; }
      conv1_wmma_k<<<dim3(8192), blk, 0, stream>>>(gin, W1P, b1, b, H1);   // 1 block/(d,h) row
      conv2_wmma_k<<<dim3(8192), blk, 0, stream>>>(H1, W2P, b2, b, H2);    // 1 block/(d,h) row
      conv3_res_k<<<dim3(V_/256), blk, 0, stream>>>(H2, w3, b3, gin, b, nouts[b]);
    }
    update_pqs_k<<<dim3(V_/256), blk, 0, stream>>>(P, Q, S, PN, QN, SN, ntx, nty, ntz, c);
    combine_k<<<dim3(V_/256), blk, 0, stream>>>(P, Q, S, t_in, ZN, ntt, c, t_out);
  }
}
